// GeometryConstraitUpdate_8899172237443
// MI455X (gfx1250) — matrix operand order repaired and checked by reading, not compiled, not run on hardware
//
#include <hip/hip_runtime.h>
#include <hip/hip_bf16.h>

typedef __bf16 bf16;
typedef __attribute__((ext_vector_type(16))) __bf16 v16bf;
typedef __attribute__((ext_vector_type(8)))  __bf16 v8bf;
typedef __attribute__((ext_vector_type(8)))  float  v8f;

constexpr int NH   = 8;       // heads
constexpr int CC   = 32;      // head dim
constexpr int DD   = 128;     // model dim
constexpr int LQ   = 256;     // sequence (L)
constexpr int NI   = 128;     // I
constexpr int HCC  = 256;     // H*C
constexpr int NROW = NI * LQ; // 32768 rows

#define USE_ASYNC_LDS 1

// Build a 16-bit A-fragment (16x32) from two contiguous 8-elem chunks:
// e<8 -> k = kh*16+e  (chunk at base + kh*16), e>=8 -> k = kh*16+8+(e-8) (chunk at base+kh*16+8)
__device__ __forceinline__ v16bf make_a_frag(const bf16* base, int kh) {
  v8bf lo = *reinterpret_cast<const v8bf*>(base + kh * 16);
  v8bf hi = *reinterpret_cast<const v8bf*>(base + kh * 16 + 8);
  v16bf a;
#pragma unroll
  for (int e = 0; e < 8; ++e) { a[e] = lo[e]; a[8 + e] = hi[e]; }
  return a;
}

__device__ __forceinline__ float bf2f(bf16 x) { return (float)x; }
__device__ __forceinline__ bf16  f2bf(float x) { return (bf16)x; }

// ---------------- weight fp32 -> bf16 cast WITH transpose ----------------
// src is [K][N] row-major fp32; dst is [N][K] bf16 so WMMA B-fragments are contiguous.
__global__ void castw_t_kernel(const float* __restrict__ src, bf16* __restrict__ dst,
                               int K, int N) {
  int t = blockIdx.x * blockDim.x + threadIdx.x;
  if (t < K * N) {
    int k = t / N, n = t % N;
    dst[(size_t)n * K + k] = f2bf(src[t]);
  }
}

// ---------------- LayerNorm (fp32) + cast to bf16 ----------------
__global__ void __launch_bounds__(256) ln_kernel(const float* __restrict__ z,
                                                 const float* __restrict__ gamma,
                                                 const float* __restrict__ beta,
                                                 bf16* __restrict__ zln) {
  const int lane = threadIdx.x & 31;
  const int row  = blockIdx.x * 8 + (threadIdx.x >> 5);
  const float4 x = reinterpret_cast<const float4*>(z + (size_t)row * DD)[lane];
  float s  = x.x + x.y + x.z + x.w;
  float sq = x.x * x.x + x.y * x.y + x.z * x.z + x.w * x.w;
#pragma unroll
  for (int off = 1; off < 32; off <<= 1) {
    s  += __shfl_xor(s,  off, 32);
    sq += __shfl_xor(sq, off, 32);
  }
  const float mu   = s * (1.0f / DD);
  const float var  = sq * (1.0f / DD) - mu * mu;
  const float rstd = rsqrtf(var + 1e-5f);
  const float v[4] = {x.x, x.y, x.z, x.w};
#pragma unroll
  for (int j = 0; j < 4; ++j) {
    int c = lane * 4 + j;
    zln[(size_t)row * DD + c] = f2bf((v[j] - mu) * rstd * gamma[c] + beta[c]);
  }
}

// ---------------- pair bias b[i,h,q] = sum_c zln[i,q,c]*Wb[c,h] ----------------
__global__ void bias_b_kernel(const bf16* __restrict__ zln, const float* __restrict__ Wb,
                              float* __restrict__ bbuf) {
  int t = blockIdx.x * blockDim.x + threadIdx.x;  // NROW*NH threads
  int row = t >> 3, h = t & 7;
  const bf16* zr = zln + (size_t)row * DD;
  float s = 0.f;
#pragma unroll
  for (int cb = 0; cb < DD; cb += 8) {
    v8bf zv = *reinterpret_cast<const v8bf*>(zr + cb);
#pragma unroll
    for (int j = 0; j < 8; ++j) s += bf2f(zv[j]) * Wb[(cb + j) * NH + h];
  }
  int i = row >> 8, q = row & 255;
  bbuf[(i * NH + h) * LQ + q] = s;
}

// ---------------- d[h,j,k] = d_jk[j,k]*Wd[h] ----------------
__global__ void bias_d_kernel(const float* __restrict__ djk, const float* __restrict__ Wd,
                              float* __restrict__ dpre) {
  int t = blockIdx.x * blockDim.x + threadIdx.x;  // NH*LQ*LQ
  int h = t >> 16, jk = t & 65535;
  dpre[t] = djk[jk] * Wd[h];
}

// ---------------- fused q/k/v/gate projection GEMM via WMMA ----------------
// Weights pre-transposed: Wt[n][k], n in [0,256), k in [0,128).
__global__ void __launch_bounds__(256) proj_kernel(
    const bf16* __restrict__ zln,
    const bf16* __restrict__ Wqt, const bf16* __restrict__ Wkt,
    const bf16* __restrict__ Wvt, const bf16* __restrict__ Wgt,
    const float* __restrict__ bg,
    bf16* __restrict__ qo, bf16* __restrict__ ko,
    bf16* __restrict__ vo, bf16* __restrict__ go) {
  const int lane = threadIdx.x & 31;
  const int wv   = threadIdx.x >> 5;
  const int hf   = lane >> 4;
  const int l15  = lane & 15;
  const int mt   = blockIdx.x;
  const int pj   = blockIdx.y;
  const bf16* W  = (pj == 0) ? Wqt : (pj == 1) ? Wkt : (pj == 2) ? Wvt : Wgt;
  bf16* out      = (pj == 0) ? qo  : (pj == 1) ? ko  : (pj == 2) ? vo  : go;
  const int n0 = wv * 32;
  v8f acc0, acc1;
#pragma unroll
  for (int r = 0; r < 8; ++r) { acc0[r] = 0.f; acc1[r] = 0.f; }
  const bf16* arow = zln + (size_t)(mt * 16 + l15) * DD;
  const bf16* wr0  = W + (size_t)(n0 + l15) * DD;       // B col n0+l15
  const bf16* wr1  = W + (size_t)(n0 + 16 + l15) * DD;  // B col n0+16+l15
#pragma unroll
  for (int kb = 0; kb < DD; kb += 32) {
    v16bf a  = make_a_frag(arow + kb, hf);
    // B-fragment: elem e -> k = hf*16+e, contiguous in transposed weights
    v16bf b0 = *reinterpret_cast<const v16bf*>(wr0 + kb + hf * 16);
    v16bf b1 = *reinterpret_cast<const v16bf*>(wr1 + kb + hf * 16);
    acc0 = __builtin_amdgcn_wmma_f32_16x16x32_bf16(false, a, false, b0, (short)0, acc0, false, false);
    acc1 = __builtin_amdgcn_wmma_f32_16x16x32_bf16(false, a, false, b1, (short)0, acc1, false, false);
  }
  const float qscale = 0.35355339059327373f;  // 8^-0.5
#pragma unroll
  for (int r = 0; r < 8; ++r) {
    int row = mt * 16 + r + 8 * hf;
    int c0 = n0 + l15, c1 = c0 + 16;
    float v0 = acc0[r], v1 = acc1[r];
    if (pj == 0) { v0 *= qscale; v1 *= qscale; }
    if (pj == 3) { v0 += bg[c0]; v1 += bg[c1]; }
    out[(size_t)row * HCC + c0] = f2bf(v0);
    out[(size_t)row * HCC + c1] = f2bf(v1);
  }
}

// ---------------- flash attention per (i,h) block ----------------
__global__ void __launch_bounds__(256) attn_kernel(
    const bf16* __restrict__ qb, const bf16* __restrict__ kbuf,
    const bf16* __restrict__ vbuf, const bf16* __restrict__ gateb,
    const float* __restrict__ bbuf, const float* __restrict__ dpre,
    const float* __restrict__ maskz, bf16* __restrict__ gw) {
  __shared__ __align__(16) bf16 Ks[LQ * CC];     // 16 KB, row-major [krow][c]
  __shared__ __align__(16) bf16 Vt[CC * LQ];     // 16 KB, transposed [c][krow]
  __shared__ __align__(16) bf16 Ps[8][16 * 32];  //  8 KB, P scratch per wave
  const int i    = blockIdx.x >> 3;
  const int h    = blockIdx.x & 7;
  const int lane = threadIdx.x & 31;
  const int w    = threadIdx.x >> 5;
  const int hf   = lane >> 4;
  const int l15  = lane & 15;
  {
    const int row = threadIdx.x;  // 256 threads <-> 256 K/V rows
    const bf16* ksrc = kbuf + (size_t)(i * LQ + row) * HCC + h * CC;
    const bf16* vsrc = vbuf + (size_t)(i * LQ + row) * HCC + h * CC;
#if USE_ASYNC_LDS
    // K rows DMA'd straight to LDS: 64B per row as 4x async b128 (ASYNCcnt path)
    {
      unsigned ldsoff = (unsigned)(uintptr_t)(&Ks[row * CC]);
      unsigned long long gaddr = (unsigned long long)(uintptr_t)ksrc;
      asm volatile(
          "global_load_async_to_lds_b128 %0, %1, off\n\t"
          "global_load_async_to_lds_b128 %0, %1, off offset:16\n\t"
          "global_load_async_to_lds_b128 %0, %1, off offset:32\n\t"
          "global_load_async_to_lds_b128 %0, %1, off offset:48"
          :: "v"(ldsoff), "v"(gaddr) : "memory");
    }
#else
    {
      const uint4* ks4 = reinterpret_cast<const uint4*>(ksrc);
      uint4* kd4 = reinterpret_cast<uint4*>(&Ks[row * CC]);
#pragma unroll
      for (int j = 0; j < 4; ++j) kd4[j] = ks4[j];
    }
#endif
    // V staged transposed (scalar scatter, one-time cost)
    const v8bf* vs8 = reinterpret_cast<const v8bf*>(vsrc);
#pragma unroll
    for (int j = 0; j < 4; ++j) {
      v8bf vv = vs8[j];
#pragma unroll
      for (int e = 0; e < 8; ++e) Vt[(j * 8 + e) * LQ + row] = vv[e];
    }
#if USE_ASYNC_LDS
    asm volatile("s_wait_asynccnt 0x0" ::: "memory");
#endif
  }
  __syncthreads();
  for (int qi = 0; qi < 2; ++qi) {
    const int qt = w + qi * 8;
    const int qbase = qt * 16;
    const bf16* qrow = qb + (size_t)(i * LQ + qbase + l15) * HCC + h * CC;
    v16bf aq = make_a_frag(qrow, hf);
    float bb[8];
#pragma unroll
    for (int r = 0; r < 8; ++r)
      bb[r] = bbuf[(i * NH + h) * LQ + qbase + r + 8 * hf];
    float mrow[8], srow[8];
    v8f acc0, acc1;
#pragma unroll
    for (int r = 0; r < 8; ++r) { mrow[r] = -1e30f; srow[r] = 0.f; acc0[r] = 0.f; acc1[r] = 0.f; }
    for (int kb = 0; kb < LQ; kb += 32) {
      // --- logits: Q(16x32) @ K^T. B elem e -> c = hf*16+e, contiguous in Ks row ---
      v16bf bk0 = *reinterpret_cast<const v16bf*>(&Ks[(kb + l15) * CC + hf * 16]);
      v16bf bk1 = *reinterpret_cast<const v16bf*>(&Ks[(kb + 16 + l15) * CC + hf * 16]);
      v8f zc;
#pragma unroll
      for (int r = 0; r < 8; ++r) zc[r] = 0.f;
      v8f s0 = __builtin_amdgcn_wmma_f32_16x16x32_bf16(false, aq, false, bk0, (short)0, zc, false, false);
      v8f s1 = __builtin_amdgcn_wmma_f32_16x16x32_bf16(false, aq, false, bk1, (short)0, zc, false, false);
      const int kc0 = kb + l15, kc1 = kc0 + 16;
      const float am0 = 1e9f * (maskz[i * LQ + kc0] - 1.0f);
      const float am1 = 1e9f * (maskz[i * LQ + kc1] - 1.0f);
#pragma unroll
      for (int r = 0; r < 8; ++r) {
        int qr = qbase + r + 8 * hf;
        const float* drow = dpre + (size_t)(h * LQ + qr) * LQ;
        s0[r] += bb[r] + drow[kc0] + am0;
        s1[r] += bb[r] + drow[kc1] + am1;
      }
      // --- streaming softmax: row-wise reductions within 16-lane halves ---
#pragma unroll
      for (int r = 0; r < 8; ++r) {
        float rm = fmaxf(s0[r], s1[r]);
#pragma unroll
        for (int off = 1; off < 16; off <<= 1) rm = fmaxf(rm, __shfl_xor(rm, off, 32));
        float nm = fmaxf(mrow[r], rm);
        float scale = __expf(mrow[r] - nm);
        mrow[r] = nm;
        float p0 = __expf(s0[r] - nm);
        float p1 = __expf(s1[r] - nm);
        s0[r] = p0; s1[r] = p1;
        float rs = p0 + p1;
#pragma unroll
        for (int off = 1; off < 16; off <<= 1) rs += __shfl_xor(rs, off, 32);
        srow[r] = srow[r] * scale + rs;
        acc0[r] *= scale;
        acc1[r] *= scale;
      }
      // --- transpose P (C-layout -> A-layout) through LDS, as bf16 ---
#pragma unroll
      for (int r = 0; r < 8; ++r) {
        int m = r + 8 * hf;
        Ps[w][m * 32 + l15]      = f2bf(s0[r]);
        Ps[w][m * 32 + 16 + l15] = f2bf(s1[r]);
      }
      __syncthreads();
      v16bf pa = make_a_frag(&Ps[w][l15 * 32], hf);
      // --- PV: B elem e -> krow = kb+hf*16+e, contiguous in transposed Vt ---
      v16bf bv0 = *reinterpret_cast<const v16bf*>(&Vt[l15 * LQ + kb + hf * 16]);
      v16bf bv1 = *reinterpret_cast<const v16bf*>(&Vt[(16 + l15) * LQ + kb + hf * 16]);
      acc0 = __builtin_amdgcn_wmma_f32_16x16x32_bf16(false, pa, false, bv0, (short)0, acc0, false, false);
      acc1 = __builtin_amdgcn_wmma_f32_16x16x32_bf16(false, pa, false, bv1, (short)0, acc1, false, false);
      __syncthreads();
    }
    // --- epilogue: normalize, gate, store bf16 ---
#pragma unroll
    for (int r = 0; r < 8; ++r) {
      float inv = 1.0f / fmaxf(srow[r], 1e-30f);
      int row = i * LQ + qbase + r + 8 * hf;
      int c0 = h * CC + l15, c1 = c0 + 16;
      float g0 = 1.0f / (1.0f + __expf(-bf2f(gateb[(size_t)row * HCC + c0])));
      float g1 = 1.0f / (1.0f + __expf(-bf2f(gateb[(size_t)row * HCC + c1])));
      gw[(size_t)row * HCC + c0] = f2bf(acc0[r] * inv * g0);
      gw[(size_t)row * HCC + c1] = f2bf(acc1[r] * inv * g1);
    }
  }
}

// ---------------- output GEMM: gw[32768,256] @ Wf[256,128] + bf, *mask ----------------
// Wft pre-transposed: [n=128][k=256]
__global__ void __launch_bounds__(256) out_gemm_kernel(
    const bf16* __restrict__ gw, const bf16* __restrict__ Wft,
    const float* __restrict__ bf_, const float* __restrict__ maskz,
    float* __restrict__ out) {
  const int lane = threadIdx.x & 31;
  const int w    = threadIdx.x >> 5;
  const int hf   = lane >> 4;
  const int l15  = lane & 15;
  const int mt   = blockIdx.x;
  const int n0   = w * 16;
  v8f acc;
#pragma unroll
  for (int r = 0; r < 8; ++r) acc[r] = 0.f;
  const bf16* arow = gw + (size_t)(mt * 16 + l15) * HCC;
  const bf16* wr   = Wft + (size_t)(n0 + l15) * HCC;
#pragma unroll
  for (int kb = 0; kb < HCC; kb += 32) {
    v16bf a = make_a_frag(arow + kb, hf);
    v16bf b = *reinterpret_cast<const v16bf*>(wr + kb + hf * 16);
    acc = __builtin_amdgcn_wmma_f32_16x16x32_bf16(false, a, false, b, (short)0, acc, false, false);
  }
#pragma unroll
  for (int r = 0; r < 8; ++r) {
    int row = mt * 16 + r + 8 * hf;
    int col = n0 + l15;
    out[(size_t)row * DD + col] = (acc[r] + bf_[col]) * maskz[row];
  }
}

extern "C" void kernel_launch(void* const* d_in, const int* in_sizes, int n_in,
                              void* d_out, int out_size, void* d_ws, size_t ws_size,
                              hipStream_t stream) {
  const float* z_ij  = (const float*)d_in[0];
  const float* d_jk  = (const float*)d_in[1];
  const float* maskz = (const float*)d_in[2];
  const float* gamma = (const float*)d_in[3];
  const float* beta  = (const float*)d_in[4];
  const float* Wq    = (const float*)d_in[5];
  const float* Wk    = (const float*)d_in[6];
  const float* Wv    = (const float*)d_in[7];
  const float* Wb    = (const float*)d_in[8];
  const float* Wd    = (const float*)d_in[9];
  const float* Wg    = (const float*)d_in[10];
  const float* bg    = (const float*)d_in[11];
  const float* Wf    = (const float*)d_in[12];
  const float* bf_   = (const float*)d_in[13];
  float* out = (float*)d_out;

  char* p = (char*)d_ws;
  auto carve = [&](size_t bytes) {
    char* r = p;
    p += (bytes + 255) & ~size_t(255);
    return r;
  };
  bf16*  zln  = (bf16*)carve((size_t)NROW * DD * 2);
  bf16*  qbuf = (bf16*)carve((size_t)NROW * HCC * 2);
  bf16*  kbuf = (bf16*)carve((size_t)NROW * HCC * 2);
  bf16*  vbuf = (bf16*)carve((size_t)NROW * HCC * 2);
  bf16*  gbuf = (bf16*)carve((size_t)NROW * HCC * 2);
  bf16*  gw   = (bf16*)carve((size_t)NROW * HCC * 2);
  float* bbuf = (float*)carve((size_t)NI * NH * LQ * 4);
  float* dpre = (float*)carve((size_t)NH * LQ * LQ * 4);
  bf16*  Wqt  = (bf16*)carve((size_t)DD * HCC * 2);
  bf16*  Wkt  = (bf16*)carve((size_t)DD * HCC * 2);
  bf16*  Wvt  = (bf16*)carve((size_t)DD * HCC * 2);
  bf16*  Wgt  = (bf16*)carve((size_t)DD * HCC * 2);
  bf16*  Wft  = (bf16*)carve((size_t)HCC * DD * 2);

  const int nw = DD * HCC;  // 32768
  castw_t_kernel<<<(nw + 255) / 256, 256, 0, stream>>>(Wq, Wqt, DD, HCC);
  castw_t_kernel<<<(nw + 255) / 256, 256, 0, stream>>>(Wk, Wkt, DD, HCC);
  castw_t_kernel<<<(nw + 255) / 256, 256, 0, stream>>>(Wv, Wvt, DD, HCC);
  castw_t_kernel<<<(nw + 255) / 256, 256, 0, stream>>>(Wg, Wgt, DD, HCC);
  castw_t_kernel<<<(nw + 255) / 256, 256, 0, stream>>>(Wf, Wft, HCC, DD);

  ln_kernel<<<NROW / 8, 256, 0, stream>>>(z_ij, gamma, beta, zln);
  bias_b_kernel<<<NROW * NH / 256, 256, 0, stream>>>(zln, Wb, bbuf);
  bias_d_kernel<<<NH * LQ * LQ / 256, 256, 0, stream>>>(d_jk, Wd, dpre);
  proj_kernel<<<dim3(NROW / 16, 4), 256, 0, stream>>>(zln, Wqt, Wkt, Wvt, Wgt, bg,
                                                      qbuf, kbuf, vbuf, gbuf);
  attn_kernel<<<NI * NH, 256, 0, stream>>>(qbuf, kbuf, vbuf, gbuf, bbuf, dpre, maskz, gw);
  out_gemm_kernel<<<NROW / 16, 256, 0, stream>>>(gw, Wft, bf_, maskz, out);
}
